// VecORTree_29875792511504
// MI455X (gfx1250) — compile-verified, matrix operand order checked by reading, not run on hardware
//
#include <hip/hip_runtime.h>

// VecORTree on MI455X (gfx1250, wave32).
//
// Reference collapses to: out[row] = any(x[row, :]) ? 1.0f : 0.0f over a
// (65536, 1024) 0/1-valued f32 matrix. Equivalent to (row_sum >= 0.5).
//
// Roofline: 256 MB read / 23.3 TB/s ~= 11 us floor; FLOPs negligible ->
// pure bandwidth kernel. We stream with b128 loads and do the horizontal
// 16-row reduction for free on the matrix pipe: v_wmma_f32_16x16x32_f16
// with an all-ones B matrix yields row sums in every column of D.
//
// A-matrix 16x32 f16 layout (ISA 7.12.2): lane L in 0..15 holds row M=L,
// K = {k0+0..7, k0+16..23}; lane L+16 holds the SAME row, K = {k0+8..15,
// k0+24..31}. So lanes L and L+16 jointly cover bytes [4*k0, 4*k0+128) of
// row L contiguously each iteration -> full 128B-line consumption per step.

typedef __attribute__((ext_vector_type(16))) _Float16 v16h;
typedef __attribute__((ext_vector_type(8)))  float    v8f;

#define N_ROWS 65536
#define N_COLS 1024
#define WAVES_PER_BLOCK 8   // 256 threads = 8 wave32
#define ROWS_PER_WAVE 16

__global__ __launch_bounds__(256)
void vec_or_tree_wmma(const float* __restrict__ x, float* __restrict__ out) {
    const int tid  = threadIdx.x;
    const int lane = tid & 31;
    const int wave = tid >> 5;
    const int half = lane >> 4;        // 0: K 0..7/16..23, 1: K 8..15/24..31
    const int r    = lane & 15;        // row within the 16-row tile

    const long rowTile = (long)blockIdx.x * WAVES_PER_BLOCK + wave;
    const long rowBase = rowTile * ROWS_PER_WAVE;
    const float* __restrict__ rowPtr = x + (rowBase + r) * (long)N_COLS;
    const int khi = half * 8;

    // B = all ones (layout-invariant since every element is identical).
    v16h ones;
#pragma unroll
    for (int i = 0; i < 16; ++i) ones[i] = (_Float16)1.0f;

    v8f acc = {};

#pragma unroll 4
    for (int k0 = 0; k0 < N_COLS; k0 += 32) {
        const float4* p = (const float4*)(rowPtr + k0 + khi);
        float4 f0 = p[0];   // K = k0+khi+0..3
        float4 f1 = p[1];   // K = k0+khi+4..7
        float4 f2 = p[4];   // K = k0+khi+16..19
        float4 f3 = p[5];   // K = k0+khi+20..23

        v16h a;
        a[0]  = (_Float16)f0.x;  a[1]  = (_Float16)f0.y;
        a[2]  = (_Float16)f0.z;  a[3]  = (_Float16)f0.w;
        a[4]  = (_Float16)f1.x;  a[5]  = (_Float16)f1.y;
        a[6]  = (_Float16)f1.z;  a[7]  = (_Float16)f1.w;
        a[8]  = (_Float16)f2.x;  a[9]  = (_Float16)f2.y;
        a[10] = (_Float16)f2.z;  a[11] = (_Float16)f2.w;
        a[12] = (_Float16)f3.x;  a[13] = (_Float16)f3.y;
        a[14] = (_Float16)f3.z;  a[15] = (_Float16)f3.w;

        // acc(16x16 f32) += A(16x32 f16) * ones(32x16 f16)
        acc = __builtin_amdgcn_wmma_f32_16x16x32_f16(
            /*neg_a=*/false, a, /*neg_b=*/false, ones,
            /*c_mod=*/(short)0, acc, /*reuse_a=*/false, /*reuse_b=*/false);
    }

    // C/D layout: lane 0 holds D[M=0..7][N=0] in VGPRs 0..7,
    //             lane 16 holds D[M=8..15][N=0]. All columns are equal.
    if (r == 0) {
        float res[8];
#pragma unroll
        for (int i = 0; i < 8; ++i) res[i] = (acc[i] >= 0.5f) ? 1.0f : 0.0f;
        float* o = out + rowBase + half * 8;
        ((float4*)o)[0] = make_float4(res[0], res[1], res[2], res[3]);
        ((float4*)o)[1] = make_float4(res[4], res[5], res[6], res[7]);
    }
}

extern "C" void kernel_launch(void* const* d_in, const int* in_sizes, int n_in,
                              void* d_out, int out_size, void* d_ws, size_t ws_size,
                              hipStream_t stream) {
    const float* x   = (const float*)d_in[0];
    float*       out = (float*)d_out;

    // 65536 rows / (16 rows per wave * 8 waves per block) = 512 blocks.
    dim3 grid(N_ROWS / (ROWS_PER_WAVE * WAVES_PER_BLOCK));
    dim3 block(WAVES_PER_BLOCK * 32);
    hipLaunchKernelGGL(vec_or_tree_wmma, grid, block, 0, stream, x, out);
}
